// MultiheadAttention_18038862643926
// MI455X (gfx1250) — matrix operand order repaired and checked by reading, not compiled, not run on hardware
//
#include <hip/hip_runtime.h>
#include <hip/hip_bf16.h>

#define DI __device__ __forceinline__

typedef __attribute__((ext_vector_type(16))) _Float16 v16h;
typedef __attribute__((ext_vector_type(8)))  float    v8f;

namespace {

constexpr int kB = 2, kT = 2048, kS = 2048, kD = 1024, kH = 16, kHD = 64;
constexpr float kScale = 0.125f;      // 1/sqrt(64)
constexpr float kInvH  = 1.0f / (float)kH;

union FragH { v16h h; _Float16 e[16]; unsigned int u[8]; };

DI int laneId() { return (int)(threadIdx.x & 31u); }

// A fragment (16x32 f16): element (m,k) from row-major p[m*ld + k].
// Layout: lane = m + 16*(k>>4); vgpr v holds k = (lane>>4)*16 + 2*v (+1).
DI v16h load_fragA(const _Float16* p, int ld) {
  int lane = laneId();
  int m  = lane & 15;
  int kh = (lane >> 4) * 16;
  const _Float16* rp = p + (size_t)m * ld + kh;
  FragH f;
#pragma unroll
  for (int v = 0; v < 8; ++v) {
    int k = 2 * v;
    f.u[v] = *(const unsigned int*)(rp + k);
  }
  return f.h;
}

// B fragment (32x16): B[k][n] = p[n*ld + k]  (operand stored row-major [N,K], e.g. W or K-matrix).
// Layout: lanes 0-15 hold K=0..15 (vgpr v: k=2v,2v+1), lanes 16-31 hold K=16..31; n = lane&15.
DI v16h load_fragB_nk(const _Float16* p, int ld) {
  int lane = laneId();
  int n   = lane & 15;
  int khi = (lane >> 4) << 4;
  const _Float16* rp = p + (size_t)n * ld + khi;
  FragH f;
#pragma unroll
  for (int v = 0; v < 8; ++v) f.u[v] = *(const unsigned int*)(rp + 2 * v);
  return f.h;
}

// B fragment (32x16): B[k][n] = p[k*ld + n]  (operand stored row-major [K,N], e.g. V-matrix).
DI v16h load_fragB_kn(const _Float16* p, int ld) {
  int lane = laneId();
  int n   = lane & 15;
  int khi = (lane >> 4) << 4;
  FragH f;
#pragma unroll
  for (int v = 0; v < 8; ++v) {
    f.e[2 * v]     = p[(size_t)(khi + 2 * v)     * ld + n];
    f.e[2 * v + 1] = p[(size_t)(khi + 2 * v + 1) * ld + n];
  }
  return f.h;
}

DI v8f wmma32(v16h a, v16h b, v8f c) {
  return __builtin_amdgcn_wmma_f32_16x16x32_f16(false, a, false, b, (short)0, c,
                                                false, false);
}

// ---------------------------------------------------------------- converts
__global__ void k_cvt(const float* __restrict__ src, _Float16* __restrict__ dst, int n) {
  int i = blockIdx.x * blockDim.x + threadIdx.x;
  if (i < n) dst[i] = (_Float16)src[i];
}

// ---------------------------------------------------------------- in-projection
// Y[i,j] = sum_d X[i,d] * W[j,d] + bias[j], j in [0,3D).
// 32x32 output per wave (2x2 WMMA tiles): 4 WMMA per 4 fragment loads per K-step,
// ~120 VGPRs total -> no spills (4x4 blocking spilled to scratch).
// sel 0/1 (Q,K): write f32 to Yqk [B*T, 2D] for the RoPE pass.
// sel 2   (V) : head-split, write f16 to Vh [B,H,S,64].
__global__ void k_inproj(const _Float16* __restrict__ Xq, const _Float16* __restrict__ Xk,
                         const _Float16* __restrict__ Xv, const _Float16* __restrict__ W,
                         const float* __restrict__ bias,
                         float* __restrict__ Yqk, _Float16* __restrict__ Vh) {
  int wid = blockIdx.x * 8 + (int)(threadIdx.x >> 5);
  const int CT = (3 * kD) / 32;                 // 96 col tiles
  int ct = wid % CT, rt = wid / CT;
  int row0 = rt * 32, col0 = ct * 32;           // 32-tiles never straddle Q/K/V thirds
  int sel = col0 / kD;
  const _Float16* X = (sel == 0) ? Xq : ((sel == 1) ? Xk : Xv);

  v8f acc00 = {}, acc01 = {}, acc10 = {}, acc11 = {};
  for (int k0 = 0; k0 < kD; k0 += 32) {
    v16h a0 = load_fragA(X + (size_t)row0 * kD + k0, kD);
    v16h a1 = load_fragA(X + (size_t)(row0 + 16) * kD + k0, kD);
    v16h b0 = load_fragB_nk(W + (size_t)col0 * kD + k0, kD);
    v16h b1 = load_fragB_nk(W + (size_t)(col0 + 16) * kD + k0, kD);
    acc00 = wmma32(a0, b0, acc00);
    acc01 = wmma32(a0, b1, acc01);
    acc10 = wmma32(a1, b0, acc10);
    acc11 = wmma32(a1, b1, acc11);
  }

  int lane = laneId(), hf = lane >> 4, n = lane & 15;
  v8f* accs[2][2] = {{&acc00, &acc01}, {&acc10, &acc11}};
  if (sel < 2) {
#pragma unroll
    for (int j = 0; j < 2; ++j) {
      float bv = bias[col0 + 16 * j + n];
#pragma unroll
      for (int i = 0; i < 2; ++i)
#pragma unroll
        for (int v = 0; v < 8; ++v) {
          int m = v + 8 * hf;
          Yqk[((size_t)row0 + 16 * i + m) * (2 * kD) + col0 + 16 * j + n] =
              (*accs[i][j])[v] + bv;
        }
    }
  } else {
#pragma unroll
    for (int j = 0; j < 2; ++j) {
      int jc = col0 + 16 * j + n - 2 * kD;
      int h = jc >> 6, hd = jc & 63;
      float bv = bias[col0 + 16 * j + n];
#pragma unroll
      for (int i = 0; i < 2; ++i)
#pragma unroll
        for (int v = 0; v < 8; ++v) {
          int row = row0 + 16 * i + v + 8 * hf;
          int b = row / kS, t = row % kS;
          Vh[(((size_t)b * kH + h) * kS + t) * kHD + hd] =
              (_Float16)((*accs[i][j])[v] + bv);
        }
    }
  }
}

// ---------------------------------------------------------------- RoPE + head-split (Q,K)
__global__ void k_rope(const float* __restrict__ Yqk,
                       _Float16* __restrict__ Qh, _Float16* __restrict__ Kh) {
  int idx = blockIdx.x * blockDim.x + threadIdx.x;
  const int tot1 = kB * kH * kT * (kHD / 2);
  if (idx >= 2 * tot1) return;
  int isK = idx >= tot1;
  int r = idx - isK * tot1;
  int hd2 = r & 31; r >>= 5;
  int t = r & (kT - 1); r >>= 11;
  int h = r & (kH - 1); int b = r >> 4;

  const float* yp = Yqk + ((size_t)b * kT + t) * (2 * kD)
                        + (size_t)isK * kD + (size_t)h * kHD + hd2;
  float x1 = yp[0], x2 = yp[32];
  // inv_freq = 10000^(-(2*hd2)/64)
  float inv = __expf(-((float)(2 * hd2) / (float)kHD) * 9.2103403719761827f);
  float ang = (float)t * inv;
  float cs = __cosf(ang), sn = __sinf(ang);
  _Float16* dst = (isK ? Kh : Qh) + (((size_t)b * kH + h) * kT + t) * kHD + hd2;
  dst[0]  = (_Float16)(x1 * cs - x2 * sn);
  dst[32] = (_Float16)(x1 * sn + x2 * cs);
}

// ---------------------------------------------------------------- flash pass 1
// One wave per (b,h,16-row Q tile): online softmax over S, O = softmax(QK^T/s)V.
// Stores O (f16, [B,T,H*64]) and per-row stats m (running max) and l (sum exp).
__global__ void k_flash(const _Float16* __restrict__ Qh, const _Float16* __restrict__ Kh,
                        const _Float16* __restrict__ Vh, const unsigned char* __restrict__ mask,
                        _Float16* __restrict__ Oh, float* __restrict__ Mrow,
                        float* __restrict__ Lrow) {
  __shared__ _Float16 plds[8][16 * 32];
  int wave = (int)(threadIdx.x >> 5);
  int lane = laneId();
  int wid = blockIdx.x * 8 + wave;
  int tt = wid & 127;                 // T/16 = 128 row tiles
  int bh = wid >> 7;                  // [0, B*H)
  int h = bh & (kH - 1), b = bh >> 4;
  int t0 = tt * 16;
  int hf = lane >> 4, n = lane & 15;

  const _Float16* qb = Qh + ((size_t)bh * kT + t0) * kHD;
  v16h aq0 = load_fragA(qb, kHD);
  v16h aq1 = load_fragA(qb + 32, kHD);

  v8f o0 = {}, o1 = {}, o2 = {}, o3 = {};
  float mrun[8], lrun[8];
#pragma unroll
  for (int v = 0; v < 8; ++v) { mrun[v] = -__builtin_inff(); lrun[v] = 0.f; }

  const unsigned char* mb = mask + (size_t)b * kS;
  for (int s0 = 0; s0 < kS; s0 += 32) {
    const _Float16* kb = Kh + ((size_t)bh * kS + s0) * kHD;
    v8f c0 = {}, c1 = {};
    c0 = wmma32(aq0, load_fragB_nk(kb, kHD), c0);
    c0 = wmma32(aq1, load_fragB_nk(kb + 32, kHD), c0);
    const _Float16* kb1 = kb + 16 * kHD;
    c1 = wmma32(aq0, load_fragB_nk(kb1, kHD), c1);
    c1 = wmma32(aq1, load_fragB_nk(kb1 + 32, kHD), c1);

    bool msk0 = mb[s0 + n] != 0;
    bool msk1 = mb[s0 + 16 + n] != 0;

#pragma unroll
    for (int v = 0; v < 8; ++v) {
      float s0v = msk0 ? -__builtin_inff() : c0[v] * kScale;
      float s1v = msk1 ? -__builtin_inff() : c1[v] * kScale;
      float mx = fmaxf(s0v, s1v);
#pragma unroll
      for (int d = 1; d < 16; d <<= 1) mx = fmaxf(mx, __shfl_xor(mx, d, 32));
      float nm = fmaxf(mrun[v], mx);
      float alpha = (mrun[v] == -__builtin_inff()) ? 0.f : __expf(mrun[v] - nm);
      float e0 = (s0v == -__builtin_inff()) ? 0.f : __expf(s0v - nm);
      float e1 = (s1v == -__builtin_inff()) ? 0.f : __expf(s1v - nm);
      float rs = e0 + e1;
#pragma unroll
      for (int d = 1; d < 16; d <<= 1) rs += __shfl_xor(rs, d, 32);
      lrun[v] = lrun[v] * alpha + rs;
      mrun[v] = nm;
      o0[v] *= alpha; o1[v] *= alpha; o2[v] *= alpha; o3[v] *= alpha;
      int m = v + 8 * hf;
      plds[wave][m * 32 + n]      = (_Float16)e0;
      plds[wave][m * 32 + 16 + n] = (_Float16)e1;
    }

    v16h pa = load_fragA(&plds[wave][0], 32);
    const _Float16* vb = Vh + ((size_t)bh * kS + s0) * kHD;
    o0 = wmma32(pa, load_fragB_kn(vb + 0,  kHD), o0);
    o1 = wmma32(pa, load_fragB_kn(vb + 16, kHD), o1);
    o2 = wmma32(pa, load_fragB_kn(vb + 32, kHD), o2);
    o3 = wmma32(pa, load_fragB_kn(vb + 48, kHD), o3);
  }

  _Float16* ob = Oh + ((size_t)b * kT + t0) * kD + (size_t)h * kHD;
#pragma unroll
  for (int v = 0; v < 8; ++v) {
    float inv = (lrun[v] > 0.f) ? 1.f / lrun[v] : 0.f;   // fully-masked row -> 0
    int m = v + 8 * hf;
    _Float16* orow = ob + (size_t)m * kD;
    orow[0 * 16 + n] = (_Float16)(o0[v] * inv);
    orow[1 * 16 + n] = (_Float16)(o1[v] * inv);
    orow[2 * 16 + n] = (_Float16)(o2[v] * inv);
    orow[3 * 16 + n] = (_Float16)(o3[v] * inv);
  }
  if (n == 0) {
#pragma unroll
    for (int v = 0; v < 8; ++v) {
      int m = v + 8 * hf;
      Mrow[(size_t)bh * kT + t0 + m] = mrun[v];
      Lrow[(size_t)bh * kT + t0 + m] = lrun[v];
    }
  }
}

// ---------------------------------------------------------------- pass 2: head-mean attention
// One wave per (b, 16x64 [t,s] tile); loops 16 heads, recomputes QK^T, uses stored stats.
// Q fragments + row stats loaded once per head, reused across 4 s-tiles (8 WMMA/head).
__global__ void k_mean(const _Float16* __restrict__ Qh, const _Float16* __restrict__ Kh,
                       const unsigned char* __restrict__ mask,
                       const float* __restrict__ Mrow, const float* __restrict__ Lrow,
                       float* __restrict__ out2) {
  int lane = laneId();
  int wid = blockIdx.x * 8 + (int)(threadIdx.x >> 5);
  int ss = wid & 31;  int r = wid >> 5;          // S/64 = 32 col tiles
  int tt = r & 127;   int b = r >> 7;            // T/16 = 128 row tiles
  int t0 = tt * 16, s0 = ss * 64;
  int hf = lane >> 4, n = lane & 15;

  bool msk[4];
#pragma unroll
  for (int j = 0; j < 4; ++j) msk[j] = mask[(size_t)b * kS + s0 + 16 * j + n] != 0;

  float acc[4][8] = {};
  for (int h = 0; h < kH; ++h) {
    int bh = b * kH + h;
    const _Float16* qb = Qh + ((size_t)bh * kT + t0) * kHD;
    v16h aq0 = load_fragA(qb, kHD);
    v16h aq1 = load_fragA(qb + 32, kHD);

    float mrv[8], lrv[8];
    const float* mr = Mrow + (size_t)bh * kT + t0;
    const float* lr = Lrow + (size_t)bh * kT + t0;
#pragma unroll
    for (int v = 0; v < 8; ++v) { mrv[v] = mr[v + 8 * hf]; lrv[v] = lr[v + 8 * hf]; }

#pragma unroll
    for (int j = 0; j < 4; ++j) {
      const _Float16* kb = Kh + ((size_t)bh * kS + s0 + 16 * j) * kHD;
      v8f c = {};
      c = wmma32(aq0, load_fragB_nk(kb, kHD), c);
      c = wmma32(aq1, load_fragB_nk(kb + 32, kHD), c);
#pragma unroll
      for (int v = 0; v < 8; ++v) {
        float p = 0.f;
        if (!msk[j] && lrv[v] > 0.f)
          p = __expf(c[v] * kScale - mrv[v]) / lrv[v];
        acc[j][v] += p * kInvH;
      }
    }
  }
#pragma unroll
  for (int j = 0; j < 4; ++j)
#pragma unroll
    for (int v = 0; v < 8; ++v) {
      int m = v + 8 * hf;
      out2[((size_t)b * kT + t0 + m) * kS + s0 + 16 * j + n] = acc[j][v];
    }
}

// ---------------------------------------------------------------- out-projection
// 32x32 output per wave (2x2 WMMA tiles), same blocking as k_inproj.
__global__ void k_outproj(const _Float16* __restrict__ Oh, const _Float16* __restrict__ Wo,
                          const float* __restrict__ bias, float* __restrict__ out) {
  int wid = blockIdx.x * 8 + (int)(threadIdx.x >> 5);
  const int CT = kD / 32;                       // 32 col tiles
  int ct = wid % CT, rt = wid / CT;
  int row0 = rt * 32, col0 = ct * 32;

  v8f acc00 = {}, acc01 = {}, acc10 = {}, acc11 = {};
  for (int k0 = 0; k0 < kD; k0 += 32) {
    v16h a0 = load_fragA(Oh + (size_t)row0 * kD + k0, kD);
    v16h a1 = load_fragA(Oh + (size_t)(row0 + 16) * kD + k0, kD);
    v16h b0 = load_fragB_nk(Wo + (size_t)col0 * kD + k0, kD);
    v16h b1 = load_fragB_nk(Wo + (size_t)(col0 + 16) * kD + k0, kD);
    acc00 = wmma32(a0, b0, acc00);
    acc01 = wmma32(a0, b1, acc01);
    acc10 = wmma32(a1, b0, acc10);
    acc11 = wmma32(a1, b1, acc11);
  }

  int lane = laneId(), hf = lane >> 4, n = lane & 15;
  v8f* accs[2][2] = {{&acc00, &acc01}, {&acc10, &acc11}};
#pragma unroll
  for (int j = 0; j < 2; ++j) {
    float bv = bias[col0 + 16 * j + n];
#pragma unroll
    for (int i = 0; i < 2; ++i)
#pragma unroll
      for (int v = 0; v < 8; ++v) {
        int m = v + 8 * hf;
        out[((size_t)row0 + 16 * i + m) * kD + col0 + 16 * j + n] =
            (*accs[i][j])[v] + bv;
      }
  }
}

} // namespace

extern "C" void kernel_launch(void* const* d_in, const int* in_sizes, int n_in,
                              void* d_out, int out_size, void* d_ws, size_t ws_size,
                              hipStream_t stream) {
  (void)in_sizes; (void)n_in; (void)out_size; (void)ws_size;
  const float* q  = (const float*)d_in[0];
  const float* k  = (const float*)d_in[1];
  const float* v  = (const float*)d_in[2];
  const unsigned char* mask = (const unsigned char*)d_in[3];
  const float* wi = (const float*)d_in[4];
  const float* bi = (const float*)d_in[5];
  const float* wo = (const float*)d_in[6];
  const float* bo = (const float*)d_in[7];

  // Workspace carve-up (~97 MB total)
  char* ws = (char*)d_ws;
  size_t off = 0;
  auto carve = [&](size_t bytes) {
    char* p = ws + off;
    off += (bytes + 255) & ~(size_t)255;
    return p;
  };
  const size_t N1 = (size_t)kB * kT * kD;                       // 4,194,304
  _Float16* Xq = (_Float16*)carve(N1 * 2);
  _Float16* Xk = (_Float16*)carve(N1 * 2);
  _Float16* Xv = (_Float16*)carve(N1 * 2);
  _Float16* Wf = (_Float16*)carve((size_t)3 * kD * kD * 2);
  _Float16* Wo = (_Float16*)carve((size_t)kD * kD * 2);
  float*    Yqk = (float*)carve((size_t)kB * kT * 2 * kD * 4);
  _Float16* Qh = (_Float16*)carve((size_t)kB * kH * kT * kHD * 2);
  _Float16* Kh = (_Float16*)carve((size_t)kB * kH * kS * kHD * 2);
  _Float16* Vh = (_Float16*)carve((size_t)kB * kH * kS * kHD * 2);
  _Float16* Oh = (_Float16*)carve(N1 * 2);
  float* Mrow = (float*)carve((size_t)kB * kH * kT * 4);
  float* Lrow = (float*)carve((size_t)kB * kH * kT * 4);

  const int TB = 256;
  // f32 -> f16 packing
  k_cvt<<<(int)((N1 + TB - 1) / TB), TB, 0, stream>>>(q, Xq, (int)N1);
  k_cvt<<<(int)((N1 + TB - 1) / TB), TB, 0, stream>>>(k, Xk, (int)N1);
  k_cvt<<<(int)((N1 + TB - 1) / TB), TB, 0, stream>>>(v, Xv, (int)N1);
  k_cvt<<<(3 * kD * kD + TB - 1) / TB, TB, 0, stream>>>(wi, Wf, 3 * kD * kD);
  k_cvt<<<(kD * kD + TB - 1) / TB, TB, 0, stream>>>(wo, Wo, kD * kD);

  // in-projection: (B*T/32)*(3D/32) = 12288 waves, 8 waves/block
  k_inproj<<<(kB * kT / 32) * (3 * kD / 32) / 8, TB, 0, stream>>>(Xq, Xk, Xv, Wf, bi, Yqk, Vh);

  // RoPE: 2 * B*H*T*(HD/2) threads
  int nrope = 2 * kB * kH * kT * (kHD / 2);
  k_rope<<<(nrope + TB - 1) / TB, TB, 0, stream>>>(Yqk, Qh, Kh);

  // flash attention: B*H*(T/16) = 4096 waves
  k_flash<<<(kB * kH * (kT / 16)) / 8, TB, 0, stream>>>(Qh, Kh, Vh, mask, Oh, Mrow, Lrow);

  float* out1 = (float*)d_out;        // [B,T,D]
  float* out2 = out1 + N1;            // [B,T,S]

  // head-mean attention: B*(T/16)*(S/64) = 8192 waves
  k_mean<<<(kB * (kT / 16) * (kS / 64)) / 8, TB, 0, stream>>>(Qh, Kh, mask, Mrow, Lrow, out2);

  // out-projection: (B*T/32)*(D/32) = 4096 waves
  k_outproj<<<((kB * kT / 32) * (kD / 32)) / 8, TB, 0, stream>>>(Oh, Wo, bo, out1);
}
